// DenseGAT_77378130805010
// MI455X (gfx1250) — matrix operand order repaired and checked by reading, not compiled, not run on hardware
//
#include <hip/hip_runtime.h>
#include <hip/hip_bf16.h>

// ---------------------------------------------------------------------------
// 2-layer GAT for MI455X (gfx1250). GEMMs via v_wmma_f32_16x16x32_bf16,
// edge softmax/aggregation via atomics. All kernels launched on `stream`.
// ---------------------------------------------------------------------------

constexpr int   kNodes    = 100000;
constexpr int   kEdges    = 1600000;
constexpr int   kIn       = 256;
constexpr int   kHid      = 64;
constexpr int   kOut      = 40;
constexpr int   kOutPad   = 48;      // pad to 3 WMMA N-tiles
constexpr float kNegSlope = 0.2f;

typedef __attribute__((ext_vector_type(16))) __bf16 v16bf;
typedef __attribute__((ext_vector_type(8)))  float  v8f;

union BFrag {
    v16bf v;
    uint4 q[2];
};

// ---------------------------------------------------------------------------
// GEMM1: z[100000x64] = feat[100000x256] @ W[256x64]   (bf16 WMMA, f32 acc)
// One wave -> 16 node rows x 64 cols (4 N-tiles). K loop: 8 steps of 32.
// ---------------------------------------------------------------------------
__global__ __launch_bounds__(256) void gat_gemm1(const float* __restrict__ feat,
                                                 const float* __restrict__ W,
                                                 float* __restrict__ z) {
    __shared__ __align__(16) unsigned short Wt[kHid * kIn];  // [col][k] bf16 bits
    const int tid = threadIdx.x;
    for (int idx = tid; idx < kIn * kHid; idx += 256) {
        const int k = idx >> 6;   // row of W (k index)
        const int n = idx & 63;   // col of W
        const __bf16 b = (__bf16)W[idx];
        Wt[n * kIn + k] = __builtin_bit_cast(unsigned short, b);
    }
    __syncthreads();

    const int wave     = tid >> 5;
    const int lane     = tid & 31;
    const int rowBlock = blockIdx.x * 8 + wave;
    if (rowBlock >= kNodes / 16) return;   // wave-uniform: EXEC stays all-1 for WMMA
    const int row0  = rowBlock * 16;
    const int row   = lane & 15;
    const int khalf = lane >> 4;
    const float* __restrict__ arow = feat + (size_t)(row0 + row) * kIn;

    v8f c0 = {}, c1 = {}, c2 = {}, c3 = {};
    for (int kk = 0; kk < kIn; kk += 32) {
        // ---- A fragment: lane holds K [kb, kb+16)
        v16bf a;
        {
            const int kb = kk + khalf * 16;
            const float4 f0 = *(const float4*)(arow + kb);
            const float4 f1 = *(const float4*)(arow + kb + 4);
            const float4 f2 = *(const float4*)(arow + kb + 8);
            const float4 f3 = *(const float4*)(arow + kb + 12);
            a[0]  = (__bf16)f0.x; a[1]  = (__bf16)f0.y; a[2]  = (__bf16)f0.z; a[3]  = (__bf16)f0.w;
            a[4]  = (__bf16)f1.x; a[5]  = (__bf16)f1.y; a[6]  = (__bf16)f1.z; a[7]  = (__bf16)f1.w;
            a[8]  = (__bf16)f2.x; a[9]  = (__bf16)f2.y; a[10] = (__bf16)f2.z; a[11] = (__bf16)f2.w;
            a[12] = (__bf16)f3.x; a[13] = (__bf16)f3.y; a[14] = (__bf16)f3.z; a[15] = (__bf16)f3.w;
        }
        // ---- B fragments: lane holds K [kb2, kb2+16) for its column
        const int kb2 = kk + khalf * 16;
        BFrag b0, b1, b2, b3;
        {
            const uint4* p0 = (const uint4*)&Wt[(0 * 16 + row) * kIn + kb2];
            b0.q[0] = p0[0]; b0.q[1] = p0[1];
            const uint4* p1 = (const uint4*)&Wt[(1 * 16 + row) * kIn + kb2];
            b1.q[0] = p1[0]; b1.q[1] = p1[1];
            const uint4* p2 = (const uint4*)&Wt[(2 * 16 + row) * kIn + kb2];
            b2.q[0] = p2[0]; b2.q[1] = p2[1];
            const uint4* p3 = (const uint4*)&Wt[(3 * 16 + row) * kIn + kb2];
            b3.q[0] = p3[0]; b3.q[1] = p3[1];
        }
        c0 = __builtin_amdgcn_wmma_f32_16x16x32_bf16(false, a, false, b0.v, (short)0, c0, false, false);
        c1 = __builtin_amdgcn_wmma_f32_16x16x32_bf16(false, a, false, b1.v, (short)0, c1, false, false);
        c2 = __builtin_amdgcn_wmma_f32_16x16x32_bf16(false, a, false, b2.v, (short)0, c2, false, false);
        c3 = __builtin_amdgcn_wmma_f32_16x16x32_bf16(false, a, false, b3.v, (short)0, c3, false, false);
    }

    // C/D layout: VGPR r -> M = khalf*8 + r ; N = lane&15 within each tile
#pragma unroll
    for (int r = 0; r < 8; ++r) {
        const int m = khalf * 8 + r;
        float* zr = z + (size_t)(row0 + m) * kHid + row;
        zr[0]  = c0[r];
        zr[16] = c1[r];
        zr[32] = c2[r];
        zr[48] = c3[r];
    }
}

// ---------------------------------------------------------------------------
// GEMM2: z[100000x40] = h[100000x64] @ W[64x40]  (W zero-padded to 48 cols)
// ---------------------------------------------------------------------------
__global__ __launch_bounds__(256) void gat_gemm2(const float* __restrict__ h,
                                                 const float* __restrict__ W,
                                                 float* __restrict__ z) {
    __shared__ __align__(16) unsigned short Wt[kOutPad * kHid];  // [col][k] bf16 bits
    const int tid = threadIdx.x;
    for (int idx = tid; idx < kOutPad * kHid; idx += 256) {
        const int n = idx >> 6;   // col (0..47)
        const int k = idx & 63;   // k   (0..63)
        const float v = (n < kOut) ? W[k * kOut + n] : 0.0f;
        const __bf16 b = (__bf16)v;
        Wt[n * kHid + k] = __builtin_bit_cast(unsigned short, b);
    }
    __syncthreads();

    const int wave     = tid >> 5;
    const int lane     = tid & 31;
    const int rowBlock = blockIdx.x * 8 + wave;
    if (rowBlock >= kNodes / 16) return;
    const int row0  = rowBlock * 16;
    const int row   = lane & 15;
    const int khalf = lane >> 4;
    const float* __restrict__ arow = h + (size_t)(row0 + row) * kHid;

    v8f c0 = {}, c1 = {}, c2 = {};
    for (int kk = 0; kk < kHid; kk += 32) {
        v16bf a;
        {
            const int kb = kk + khalf * 16;
            const float4 f0 = *(const float4*)(arow + kb);
            const float4 f1 = *(const float4*)(arow + kb + 4);
            const float4 f2 = *(const float4*)(arow + kb + 8);
            const float4 f3 = *(const float4*)(arow + kb + 12);
            a[0]  = (__bf16)f0.x; a[1]  = (__bf16)f0.y; a[2]  = (__bf16)f0.z; a[3]  = (__bf16)f0.w;
            a[4]  = (__bf16)f1.x; a[5]  = (__bf16)f1.y; a[6]  = (__bf16)f1.z; a[7]  = (__bf16)f1.w;
            a[8]  = (__bf16)f2.x; a[9]  = (__bf16)f2.y; a[10] = (__bf16)f2.z; a[11] = (__bf16)f2.w;
            a[12] = (__bf16)f3.x; a[13] = (__bf16)f3.y; a[14] = (__bf16)f3.z; a[15] = (__bf16)f3.w;
        }
        const int kb2 = kk + khalf * 16;
        BFrag b0, b1, b2;
        {
            const uint4* p0 = (const uint4*)&Wt[(0 * 16 + row) * kHid + kb2];
            b0.q[0] = p0[0]; b0.q[1] = p0[1];
            const uint4* p1 = (const uint4*)&Wt[(1 * 16 + row) * kHid + kb2];
            b1.q[0] = p1[0]; b1.q[1] = p1[1];
            const uint4* p2 = (const uint4*)&Wt[(2 * 16 + row) * kHid + kb2];
            b2.q[0] = p2[0]; b2.q[1] = p2[1];
        }
        c0 = __builtin_amdgcn_wmma_f32_16x16x32_bf16(false, a, false, b0.v, (short)0, c0, false, false);
        c1 = __builtin_amdgcn_wmma_f32_16x16x32_bf16(false, a, false, b1.v, (short)0, c1, false, false);
        c2 = __builtin_amdgcn_wmma_f32_16x16x32_bf16(false, a, false, b2.v, (short)0, c2, false, false);
    }

#pragma unroll
    for (int r = 0; r < 8; ++r) {
        const int m = khalf * 8 + r;
        float* zr = z + (size_t)(row0 + m) * kOut + row;
        zr[0]  = c0[r];
        zr[16] = c1[r];
        if (row < 8) zr[32] = c2[r];   // cols 32..39 only
    }
}

// ---------------------------------------------------------------------------
// Node attention logits: el[i] = z[i,:].al ; er[i] = z[i,:].ar
// ---------------------------------------------------------------------------
template <int F>
__global__ void gat_attn(const float* __restrict__ z, const float* __restrict__ al,
                         const float* __restrict__ ar, float* __restrict__ el,
                         float* __restrict__ er, int n) {
    const int i = blockIdx.x * blockDim.x + threadIdx.x;
    if (i >= n) return;
    const float* zr = z + (size_t)i * F;
    float a = 0.f, b = 0.f;
#pragma unroll
    for (int f4 = 0; f4 < F / 4; ++f4) {
        const float4 v = *(const float4*)(zr + f4 * 4);
        const float4 l = *(const float4*)(al + f4 * 4);
        const float4 r = *(const float4*)(ar + f4 * 4);
        a += v.x * l.x + v.y * l.y + v.z * l.z + v.w * l.w;
        b += v.x * r.x + v.y * r.y + v.z * r.z + v.w * r.w;
    }
    el[i] = a;
    er[i] = b;
}

__global__ void fill_u32(unsigned int* __restrict__ p, unsigned int v, int n) {
    const int i = blockIdx.x * blockDim.x + threadIdx.x;
    if (i < n) p[i] = v;
}

__global__ void fill_u32x4(uint4* __restrict__ p, unsigned int v, int n4) {
    const int i = blockIdx.x * blockDim.x + threadIdx.x;
    if (i < n4) p[i] = make_uint4(v, v, v, v);
}

// ordered-uint encoding for float atomicMax: monotone in float order
__device__ __forceinline__ unsigned int ord_enc(float x) {
    const unsigned int u = __float_as_uint(x);
    return (u & 0x80000000u) ? ~u : (u | 0x80000000u);
}
__device__ __forceinline__ float ord_dec(unsigned int o) {
    const unsigned int u = (o & 0x80000000u) ? (o & 0x7FFFFFFFu) : ~o;
    return __uint_as_float(u);
}

// e[j] = leaky_relu(el[src]+er[dst]); segment-max into mbuf[dst]
__global__ void gat_edge_logit(const float* __restrict__ el, const float* __restrict__ er,
                               const int* __restrict__ src, const int* __restrict__ dst,
                               float* __restrict__ e, unsigned int* __restrict__ mbuf) {
    const int j = blockIdx.x * blockDim.x + threadIdx.x;
    if (j >= kEdges) return;
    const int s = src[j], d = dst[j];
    float v = el[s] + er[d];
    v = (v > 0.f) ? v : kNegSlope * v;
    e[j] = v;
    atomicMax(&mbuf[d], ord_enc(v));
}

// w[j] = exp(e[j]-m[dst]); segment-sum into sbuf[dst]
__global__ void gat_edge_exp(float* __restrict__ e, const int* __restrict__ dst,
                             const unsigned int* __restrict__ mbuf, float* __restrict__ sbuf) {
    const int j = blockIdx.x * blockDim.x + threadIdx.x;
    if (j >= kEdges) return;
    const int d = dst[j];
    const float w = __expf(e[j] - ord_dec(mbuf[d]));
    e[j] = w;
    atomicAdd(&sbuf[d], w);
}

// alpha[j] = w[j]/s[dst]
__global__ void gat_edge_alpha(float* __restrict__ w, const int* __restrict__ dst,
                               const float* __restrict__ sbuf) {
    const int j = blockIdx.x * blockDim.x + threadIdx.x;
    if (j >= kEdges) return;
    w[j] = w[j] / sbuf[dst[j]];
}

// agg[dst,:] += alpha[j] * z[src,:]  -- one thread per (edge, 4 features):
// one src/dst/alpha load + one float4 gather amortized over 4 atomics.
template <int F>
__global__ void gat_aggregate(const float* __restrict__ alpha, const int* __restrict__ src,
                              const int* __restrict__ dst, const float* __restrict__ z,
                              float* __restrict__ agg) {
    constexpr int G = F / 4;                       // float4 groups per row
    const long long idx = (long long)blockIdx.x * blockDim.x + threadIdx.x;
    if (idx >= (long long)kEdges * G) return;
    const int j  = (int)(idx / G);
    const int g  = (int)(idx % G);
    const int s  = src[j];
    const int d  = dst[j];
    const float a = alpha[j];
    const float4 v = *(const float4*)(z + (size_t)s * F + g * 4);
    float* o = agg + (size_t)d * F + g * 4;
    atomicAdd(o + 0, a * v.x);
    atomicAdd(o + 1, a * v.y);
    atomicAdd(o + 2, a * v.z);
    atomicAdd(o + 3, a * v.w);
}

template <int F>
__global__ void gat_bias_relu(float* __restrict__ h, const float* __restrict__ b, int total) {
    const int idx = blockIdx.x * blockDim.x + threadIdx.x;
    if (idx >= total) return;
    const float v = h[idx] + b[idx % F];
    h[idx] = (v > 0.f) ? v : 0.f;
}

// out[i,:] = log_softmax(out[i,:] + b2)  (in place, row per thread)
__global__ void gat_final(float* __restrict__ out, const float* __restrict__ b2, int n) {
    const int i = blockIdx.x * blockDim.x + threadIdx.x;
    if (i >= n) return;
    float* row = out + (size_t)i * kOut;
    float v[kOut];
    float mx = -3.0e38f;
#pragma unroll
    for (int f = 0; f < kOut; ++f) {
        v[f] = row[f] + b2[f];
        mx = fmaxf(mx, v[f]);
    }
    float s = 0.f;
#pragma unroll
    for (int f = 0; f < kOut; ++f) s += __expf(v[f] - mx);
    const float ls = __logf(s) + mx;
#pragma unroll
    for (int f = 0; f < kOut; ++f) row[f] = v[f] - ls;
}

// ---------------------------------------------------------------------------
extern "C" void kernel_launch(void* const* d_in, const int* in_sizes, int n_in,
                              void* d_out, int out_size, void* d_ws, size_t ws_size,
                              hipStream_t stream) {
    const float* feat = (const float*)d_in[0];
    const int*   src  = (const int*)d_in[1];
    const int*   dst  = (const int*)d_in[2];
    const float* W1   = (const float*)d_in[3];
    const float* b1   = (const float*)d_in[4];
    const float* al1  = (const float*)d_in[5];
    const float* ar1  = (const float*)d_in[6];
    const float* W2   = (const float*)d_in[7];
    const float* b2   = (const float*)d_in[8];
    const float* al2  = (const float*)d_in[9];
    const float* ar2  = (const float*)d_in[10];
    float*       out  = (float*)d_out;

    // workspace layout (floats)
    float* ws = (float*)d_ws;
    float* z1 = ws;                                   // N*64
    float* h1 = z1 + (size_t)kNodes * kHid;           // N*64 (agg1 -> relu'd hidden)
    float* z2 = h1 + (size_t)kNodes * kHid;           // N*40
    float* el = z2 + (size_t)kNodes * kOut;           // N
    float* er = el + kNodes;                          // N
    unsigned int* mbuf = (unsigned int*)(er + kNodes);// N
    float* sbuf = (float*)(mbuf + kNodes);            // N
    float* ebuf = sbuf + kNodes;                      // E (edge logits -> weights -> alpha)

    const int TB       = 256;
    const int gemmGrid = (kNodes / 16 + 7) / 8;                       // 782
    const int nodeGrid = (kNodes + TB - 1) / TB;
    const int edgeGrid = (kEdges + TB - 1) / TB;
    const int hGrid4   = (kNodes * kHid / 4 + TB - 1) / TB;
    const int oGrid4   = (kNodes * kOut / 4 + TB - 1) / TB;
    const int hGrid    = (kNodes * kHid + TB - 1) / TB;
    const int aggGrid1 = (int)(((long long)kEdges * (kHid / 4) + TB - 1) / TB);
    const int aggGrid2 = (int)(((long long)kEdges * (kOut / 4) + TB - 1) / TB);
    const unsigned int NEG_INF_ORD = 0x007FFFFFu;     // ord_enc(-inf)

    // ---------------- Layer 1 ----------------
    gat_gemm1<<<gemmGrid, TB, 0, stream>>>(feat, W1, z1);
    gat_attn<kHid><<<nodeGrid, TB, 0, stream>>>(z1, al1, ar1, el, er, kNodes);
    fill_u32<<<nodeGrid, TB, 0, stream>>>(mbuf, NEG_INF_ORD, kNodes);
    fill_u32<<<nodeGrid, TB, 0, stream>>>((unsigned int*)sbuf, 0u, kNodes);
    fill_u32x4<<<hGrid4, TB, 0, stream>>>((uint4*)h1, 0u, kNodes * kHid / 4);
    gat_edge_logit<<<edgeGrid, TB, 0, stream>>>(el, er, src, dst, ebuf, mbuf);
    gat_edge_exp<<<edgeGrid, TB, 0, stream>>>(ebuf, dst, mbuf, sbuf);
    gat_edge_alpha<<<edgeGrid, TB, 0, stream>>>(ebuf, dst, sbuf);
    gat_aggregate<kHid><<<aggGrid1, TB, 0, stream>>>(ebuf, src, dst, z1, h1);
    gat_bias_relu<kHid><<<hGrid, TB, 0, stream>>>(h1, b1, kNodes * kHid);

    // ---------------- Layer 2 ----------------
    gat_gemm2<<<gemmGrid, TB, 0, stream>>>(h1, W2, z2);
    gat_attn<kOut><<<nodeGrid, TB, 0, stream>>>(z2, al2, ar2, el, er, kNodes);
    fill_u32<<<nodeGrid, TB, 0, stream>>>(mbuf, NEG_INF_ORD, kNodes);
    fill_u32<<<nodeGrid, TB, 0, stream>>>((unsigned int*)sbuf, 0u, kNodes);
    fill_u32x4<<<oGrid4, TB, 0, stream>>>((uint4*)out, 0u, kNodes * kOut / 4);
    gat_edge_logit<<<edgeGrid, TB, 0, stream>>>(el, er, src, dst, ebuf, mbuf);
    gat_edge_exp<<<edgeGrid, TB, 0, stream>>>(ebuf, dst, mbuf, sbuf);
    gat_edge_alpha<<<edgeGrid, TB, 0, stream>>>(ebuf, dst, sbuf);
    gat_aggregate<kOut><<<aggGrid2, TB, 0, stream>>>(ebuf, src, dst, z2, out);
    gat_final<<<nodeGrid, TB, 0, stream>>>(out, b2, kNodes);
}
